// CNNGNNModel_40252433498617
// MI455X (gfx1250) — matrix operand order repaired and checked by reading, not compiled, not run on hardware
//
#include <hip/hip_runtime.h>
#include <hip/hip_bf16.h>
#include <math.h>

// ---------------------------------------------------------------------------
// Types for CDNA5 WMMA (wave32): bf16 A/B fragments, f32 accumulator.
// ---------------------------------------------------------------------------
typedef __attribute__((ext_vector_type(16))) __bf16   v16bf;
typedef __attribute__((ext_vector_type(8)))  __bf16   v8bf;
typedef __attribute__((ext_vector_type(8)))  float    v8f;
typedef __attribute__((ext_vector_type(4)))  unsigned u4v;
// dword-aligned float4: lets LLVM emit global_load_b128 on 4B-aligned addresses
typedef __attribute__((ext_vector_type(4), aligned(4))) float f4a;

// round-to-nearest-even f32->bf16 for two values, packed into one dword.
// Pure 32-bit integer ops: no 16-bit sub-register shuffling in codegen.
__device__ __forceinline__ unsigned pkbf16(float a, float b) {
  unsigned ua = __builtin_bit_cast(unsigned, a);
  unsigned ub = __builtin_bit_cast(unsigned, b);
  ua += 0x7FFFu + ((ua >> 16) & 1u);
  ub += 0x7FFFu + ((ub >> 16) & 1u);
  return (ua >> 16) | (ub & 0xFFFF0000u);
}

__device__ __forceinline__ v16bf frag16(const __bf16* lo, const __bf16* hi) {
  v8bf l = *(const v8bf*)lo;
  v8bf h = *(const v8bf*)hi;
  v16bf r;
#pragma unroll
  for (int i = 0; i < 8; ++i) { r[i] = l[i]; r[i + 8] = h[i]; }
  return r;
}

#define TM 128
#define TN 128
#define TK 32
#define LPAD (TK + 8)   // 80B row stride in LDS: keeps 16B frag loads bank-friendly

// ---------------------------------------------------------------------------
// Generalized strided-batched GEMM:  C[b] = A[b] * B[b] (+ bias per row)
//   A element (m,k) at A + b*saB + m*saM + k*saK   (AU=1 -> saK==1 fast path)
//   B element (k,n) at B + b*sbB + k*sbK + n*sbN   (BU=1 -> sbN==1 fast path)
//   C element (m,n) at C + b*scB + m*ldc + n
// f32 in memory, bf16 tiles in LDS, f32 accumulation via v_wmma_f32_16x16x32_bf16.
// Block = 256 threads = 8 waves; wave grid 2(M) x 4(N); wave tile 64x32 = 4x2 frags.
// Double-buffered LDS software pipeline: one barrier per K-step; next tile's
// global loads (b128 in unit-stride path) overlap the current tile's WMMAs.
// A staged as 2x b128 packed stores; B staged as k-pair packed b32 stores.
// ---------------------------------------------------------------------------
template<int AU, int BU>
__global__ __launch_bounds__(256)
void wmma_gemm(const float* __restrict__ A, long saM, long saK, long saB,
               const float* __restrict__ Bm, long sbK, long sbN, long sbB,
               float* __restrict__ C, long ldc, long scB,
               int M, int N, int K, const float* __restrict__ bias)
{
  __shared__ __bf16 As[2][TM][LPAD];
  __shared__ __bf16 Bs[2][TN][LPAD];   // stored transposed: Bs[p][n][k]

  const int bb = blockIdx.z;
  A  += (long)bb * saB;
  Bm += (long)bb * sbB;
  C  += (long)bb * scB;

  const int tid   = threadIdx.x;
  const int lane  = tid & 31;
  const int wave  = tid >> 5;
  const int waveM = (wave >> 2) * 64;   // 0 / 64
  const int waveN = (wave & 3) * 32;    // 0..96
  const int half  = lane >> 4;          // K-half selector per ISA layout
  const int lr    = lane & 15;

  const int m0 = blockIdx.y * TM;
  const int n0 = blockIdx.x * TN;

  const int ar  = tid >> 1;             // A stage: row 0..127
  const int ac  = (tid & 1) * 16;       //          16 consecutive K each
  const int bkp = tid >> 4;             // B stage: k-pair 0..15 (k = 2*bkp)
  const int bn8 = (tid & 15) * 8;       //          8 consecutive N each

  const bool aRowOk   = (m0 + ar) < M;
  const bool bColFull = (n0 + bn8 + 8) <= N;

  auto loadA = [&](int kt, float* buf) {
    const float* ap = A + (long)(m0 + ar) * saM + (long)(kt + ac) * saK;
    if (aRowOk && (kt + ac + 16) <= K) {
      if (AU) {
        const f4a* v = (const f4a*)ap;
#pragma unroll
        for (int q = 0; q < 4; ++q) {
          f4a t = v[q];
#pragma unroll
          for (int e = 0; e < 4; ++e) buf[4 * q + e] = t[e];
        }
      } else {
#pragma unroll
        for (int j = 0; j < 16; ++j) buf[j] = ap[(long)j * saK];
      }
    } else {
#pragma unroll
      for (int j = 0; j < 16; ++j)
        buf[j] = (aRowOk && (kt + ac + j) < K) ? ap[(long)j * saK] : 0.0f;
    }
  };
  // buf[0..7] = row k, buf[8..15] = row k+1 (8 consecutive columns)
  auto loadB = [&](int kt, float* buf) {
    const int k1 = kt + 2 * bkp;
    const float* bp = Bm + (long)k1 * sbK + (long)(n0 + bn8) * sbN;
    const bool ok0 = k1 < K, ok1 = (k1 + 1) < K;
    if (ok0 && ok1 && bColFull) {
      if (BU) {
        const f4a* v0 = (const f4a*)bp;
        const f4a* v1 = (const f4a*)(bp + sbK);
        f4a t0 = v0[0], t1 = v0[1], t2 = v1[0], t3 = v1[1];
#pragma unroll
        for (int e = 0; e < 4; ++e) {
          buf[e] = t0[e]; buf[4 + e]  = t1[e];
          buf[8 + e] = t2[e]; buf[12 + e] = t3[e];
        }
      } else {
#pragma unroll
        for (int j = 0; j < 8; ++j) {
          buf[j]     = bp[(long)j * sbN];
          buf[8 + j] = bp[sbK + (long)j * sbN];
        }
      }
    } else {
#pragma unroll
      for (int j = 0; j < 8; ++j) {
        bool cOk = (n0 + bn8 + j) < N;
        buf[j]     = (ok0 && cOk) ? bp[(long)j * sbN] : 0.0f;
        buf[8 + j] = (ok1 && cOk) ? bp[sbK + (long)j * sbN] : 0.0f;
      }
    }
  };
  auto storeA = [&](int p, const float* buf) {
    u4v q0, q1;
#pragma unroll
    for (int j = 0; j < 4; ++j) {
      q0[j] = pkbf16(buf[2 * j],     buf[2 * j + 1]);
      q1[j] = pkbf16(buf[8 + 2 * j], buf[9 + 2 * j]);
    }
    *(u4v*)&As[p][ar][ac]     = q0;   // 16B-aligned: p*10240 + 80*ar + {0,32}
    *(u4v*)&As[p][ar][ac + 8] = q1;
  };
  auto storeB = [&](int p, const float* buf) {
#pragma unroll
    for (int j = 0; j < 8; ++j)   // rows k,k+1 adjacent in Bs[n][k]: pack pair
      *(unsigned*)&Bs[p][bn8 + j][2 * bkp] = pkbf16(buf[j], buf[8 + j]);
  };

  v8f acc[4][2];
#pragma unroll
  for (int i = 0; i < 4; ++i)
#pragma unroll
    for (int j = 0; j < 2; ++j)
#pragma unroll
      for (int e = 0; e < 8; ++e) acc[i][j][e] = 0.0f;

  float bufA[16], bufB[16];
  loadA(0, bufA); loadB(0, bufB);
  storeA(0, bufA); storeB(0, bufB);
  __syncthreads();

  int p = 0;
  for (int k0 = 0; k0 < K; k0 += TK) {
    const bool hasNext = (k0 + TK) < K;
    if (hasNext) {               // global loads overlap WMMAs below
      loadA(k0 + TK, bufA);
      loadB(k0 + TK, bufB);
      if (AU && aRowOk && (k0 + 2 * TK) < K)   // prefetch tile after next
        __builtin_prefetch(&A[(long)(m0 + ar) * saM + (long)(k0 + 2 * TK + ac)], 0, 1);
    }

    // ---- fragment loads per ISA 16-bit layouts, then 8 WMMAs on buffer p
    v16bf af[4], bfr[2];
#pragma unroll
    for (int i = 0; i < 4; ++i) {
      const __bf16* q = &As[p][waveM + 16 * i + lr][half * 8];   // K half*8..+7 | +16
      af[i] = frag16(q, q + 16);
    }
#pragma unroll
    for (int j = 0; j < 2; ++j) {
      const __bf16* q = &Bs[p][waveN + 16 * j + lr][half * 8];   // K half*8..+7 | +16
      bfr[j] = frag16(q, q + 16);
    }
#pragma unroll
    for (int i = 0; i < 4; ++i)
#pragma unroll
      for (int j = 0; j < 2; ++j)
        acc[i][j] = __builtin_amdgcn_wmma_f32_16x16x32_bf16(
            false, af[i], false, bfr[j], (short)0, acc[i][j], false, false);

    if (hasNext) {               // fill the other buffer, then one barrier
      storeA(p ^ 1, bufA);
      storeB(p ^ 1, bufB);
    }
    __syncthreads();
    p ^= 1;
  }

  // ---- epilogue: C/D layout -> lane half gives M+8, lr gives N
#pragma unroll
  for (int i = 0; i < 4; ++i)
#pragma unroll
    for (int j = 0; j < 2; ++j)
#pragma unroll
      for (int v = 0; v < 8; ++v) {
        int gm = m0 + waveM + 16 * i + half * 8 + v;
        int gn = n0 + waveN + 16 * j + lr;
        if (gm < M && gn < N) {
          float val = acc[i][j][v];
          if (bias) val += bias[gm];
          C[(long)gm * ldc + gn] = val;
        }
      }
}

// ---------------------------------------------------------------------------
// im2col: col[b][c*kh*kw + ky*kw + kx][ho*Wo+wo]
// ---------------------------------------------------------------------------
__global__ void im2col_k(const float* __restrict__ x, float* __restrict__ col,
                         int Cin, int H, int W, int kh, int kw,
                         int stride, int pad, int Ho, int Wo)
{
  long b = blockIdx.z;
  long K = (long)Cin * kh * kw, S = (long)Ho * Wo;
  const float* xb = x + b * (long)Cin * H * W;
  float* cb = col + b * K * S;
  for (long i = blockIdx.x * (long)blockDim.x + threadIdx.x; i < K * S;
       i += (long)gridDim.x * blockDim.x) {
    long s = i % S, kidx = i / S;
    int wo = (int)(s % Wo), ho = (int)(s / Wo);
    int kx = (int)(kidx % kw);
    long t = kidx / kw;
    int ky = (int)(t % kh), c = (int)(t / kh);
    int hi = ho * stride - pad + ky, wi = wo * stride - pad + kx;
    float v = 0.0f;
    if (hi >= 0 && hi < H && wi >= 0 && wi < W) v = xb[((long)c * H + hi) * W + wi];
    cb[i] = v;
  }
}

// ---------------------------------------------------------------------------
// Training-mode BatchNorm: per-channel stats over (batch, spatial)
// ---------------------------------------------------------------------------
__global__ void bn_stats(const float* __restrict__ x, int Bn, int C, long S,
                         float* __restrict__ mean, float* __restrict__ rstd)
{
  __shared__ float sh[256], sh2[256];
  int c = blockIdx.x;
  float s = 0.f, s2 = 0.f;
  long tot = (long)Bn * S;
  for (long i = threadIdx.x; i < tot; i += blockDim.x) {
    long b = i / S, sp = i - b * S;
    float v = x[(b * C + c) * S + sp];
    s += v; s2 += v * v;
  }
  sh[threadIdx.x] = s; sh2[threadIdx.x] = s2;
  __syncthreads();
  for (int off = 128; off > 0; off >>= 1) {
    if (threadIdx.x < off) {
      sh[threadIdx.x]  += sh[threadIdx.x + off];
      sh2[threadIdx.x] += sh2[threadIdx.x + off];
    }
    __syncthreads();
  }
  if (threadIdx.x == 0) {
    float inv = 1.0f / (float)tot;
    float m = sh[0] * inv;
    float var = sh2[0] * inv - m * m;
    mean[c] = m;
    rstd[c] = rsqrtf(var + 1e-5f);
  }
}

// act: 0 none, 1 relu, 2 gelu(erf);  y = act( bn(x) + resid )
// grid: (spatial blocks, C, B) -> no per-element div/mod
__global__ void bn_apply(const float* __restrict__ x, float* __restrict__ y,
                         const float* __restrict__ mean, const float* __restrict__ rstd,
                         const float* __restrict__ g, const float* __restrict__ bt,
                         const float* __restrict__ resid,
                         int C, long S, int act)
{
  int c = blockIdx.y;
  long base = ((long)blockIdx.z * C + c) * S;
  float mu = mean[c], rs = rstd[c], ga = g[c], be = bt[c];
  for (long i = blockIdx.x * (long)blockDim.x + threadIdx.x; i < S;
       i += (long)gridDim.x * blockDim.x) {
    float v = (x[base + i] - mu) * rs * ga + be;
    if (resid) v += resid[base + i];
    if (act == 1)      v = fmaxf(v, 0.0f);
    else if (act == 2) v = 0.5f * v * (1.0f + erff(v * 0.70710678118654752f));
    y[base + i] = v;
  }
}

__global__ void maxpool3x3s2(const float* __restrict__ x, float* __restrict__ y,
                             int BC, int H, int W, int Ho, int Wo)
{
  long tot = (long)BC * Ho * Wo;
  for (long i = blockIdx.x * (long)blockDim.x + threadIdx.x; i < tot;
       i += (long)gridDim.x * blockDim.x) {
    int wo = (int)(i % Wo);
    long t = i / Wo;
    int ho = (int)(t % Ho);
    long bc = t / Ho;
    const float* p = x + bc * (long)H * W;
    float m = -3.4e38f;
    for (int dy = 0; dy < 3; ++dy) {
      int hy = ho * 2 - 1 + dy;
      if (hy < 0 || hy >= H) continue;
      for (int dx = 0; dx < 3; ++dx) {
        int wx = wo * 2 - 1 + dx;
        if (wx < 0 || wx >= W) continue;
        m = fmaxf(m, p[(long)hy * W + wx]);
      }
    }
    y[i] = m;
  }
}

// kNN (K=16, loop=True) from Gram matrix G = Y^T Y : dist = Gnn + Gmm - 2Gnm
__global__ void knn_topk(const float* __restrict__ G, int Nn,
                         int* __restrict__ idx, int total)
{
  int t = blockIdx.x * blockDim.x + threadIdx.x;
  if (t >= total) return;
  int b = t / Nn, n = t % Nn;
  const float* Gb = G + (long)b * Nn * Nn;
  float row[200];
  float dn = Gb[(long)n * Nn + n];
  for (int m = 0; m < Nn; ++m)
    row[m] = dn + Gb[(long)m * Nn + m] - 2.0f * Gb[(long)n * Nn + m];
  for (int k = 0; k < 16; ++k) {
    int bi = 0; float bv = row[0];
    for (int m = 1; m < Nn; ++m)
      if (row[m] < bv) { bv = row[m]; bi = m; }
    idx[((long)b * Nn + n) * 16 + k] = bi;
    row[bi] = 3.4e38f;
  }
}

// a_s / a_d attention scores: per (b,n,h) dot over D
__global__ void gat_scores(const float* __restrict__ h, const float* __restrict__ a_src,
                           const float* __restrict__ a_dst, float* __restrict__ as_,
                           float* __restrict__ ad_, int total4, int D)
{
  int t = blockIdx.x * blockDim.x + threadIdx.x;
  if (t >= total4) return;
  int hh = t & 3;
  long bn = t >> 2;
  const float* hp = h + bn * 4L * D + (long)hh * D;
  float s = 0.f, d = 0.f;
  for (int i = 0; i < D; ++i) {
    float v = hp[i];
    s += v * a_src[hh * D + i];
    d += v * a_dst[hh * D + i];
  }
  as_[bn * 4 + hh] = s;
  ad_[bn * 4 + hh] = d;
}

// leaky-relu + softmax over K=16 neighbors + weighted sum + head-mean + bias
// out written transposed: [B][D][Nn]
__global__ void gat_aggregate(const float* __restrict__ h, const int* __restrict__ idx,
                              const float* __restrict__ as_, const float* __restrict__ ad_,
                              const float* __restrict__ bias, float* __restrict__ out,
                              int Nn, int D)
{
  __shared__ float att[16][4];
  __shared__ int sid[16];
  int n = blockIdx.x, b = blockIdx.y;
  long bn = (long)b * Nn + n;
  if (threadIdx.x < 16) sid[threadIdx.x] = idx[bn * 16 + threadIdx.x];
  __syncthreads();
  if (threadIdx.x < 64) {
    int k = threadIdx.x >> 2, hh = threadIdx.x & 3;
    long src = (long)b * Nn + sid[k];
    float e = as_[src * 4 + hh] + ad_[bn * 4 + hh];
    att[k][hh] = (e > 0.f) ? e : 0.2f * e;
  }
  __syncthreads();
  if (threadIdx.x < 4) {
    int hh = threadIdx.x;
    float mx = -3.4e38f;
    for (int k = 0; k < 16; ++k) mx = fmaxf(mx, att[k][hh]);
    float sum = 0.f;
    for (int k = 0; k < 16; ++k) { float e = __expf(att[k][hh] - mx); att[k][hh] = e; sum += e; }
    float inv = 1.0f / sum;
    for (int k = 0; k < 16; ++k) att[k][hh] *= inv;
  }
  __syncthreads();
  for (int d = threadIdx.x; d < D; d += blockDim.x) {
    float acc = 0.f;
    for (int k = 0; k < 16; ++k) {
      const float* hp = h + ((long)b * Nn + sid[k]) * 4L * D + d;
      acc += att[k][0] * hp[0] + att[k][1] * hp[D]
           + att[k][2] * hp[2L * D] + att[k][3] * hp[3L * D];
    }
    out[((long)b * D + d) * Nn + n] = 0.25f * acc + bias[d];
  }
}

__global__ void mean_hw(const float* __restrict__ x, float* __restrict__ v,
                        int total, int S)
{
  int t = blockIdx.x * blockDim.x + threadIdx.x;
  if (t >= total) return;
  const float* p = x + (long)t * S;
  float s = 0.f;
  for (int i = 0; i < S; ++i) s += p[i];
  v[t] = s / (float)S;
}

__global__ void head_k(const float* __restrict__ v, const float* __restrict__ w,
                       const float* __restrict__ b0, float* __restrict__ out, int C)
{
  __shared__ float sh[256];
  int b = blockIdx.x;
  float s = 0.f;
  for (int c = threadIdx.x; c < C; c += 256) s += v[(long)b * C + c] * w[c];
  sh[threadIdx.x] = s;
  __syncthreads();
  for (int off = 128; off > 0; off >>= 1) {
    if (threadIdx.x < off) sh[threadIdx.x] += sh[threadIdx.x + off];
    __syncthreads();
  }
  if (threadIdx.x == 0) out[b] = sh[0] + b0[0];
}

// ---------------------------------------------------------------------------
// Host-side orchestration
// ---------------------------------------------------------------------------
namespace {

inline long cdiv(long a, long b) { return (a + b - 1) / b; }
inline int gblocks(long tot) { long g = cdiv(tot, 256); return (int)(g > 16384 ? 16384 : g); }

struct Ctx {
  void* const* din;
  int pi;
  const float* next() { return (const float*)din[pi++]; }
};

struct Arena {
  char* base; size_t off;
  float* alloc(size_t nfloats) {
    size_t b = (nfloats * sizeof(float) + 255) & ~(size_t)255;
    float* p = (float*)(base + off);
    off += b;
    return p;
  }
  void reset() { off = 0; }
};

void launch_gemm(hipStream_t st, const float* A, long saM, long saK, long saB,
                 const float* B, long sbK, long sbN, long sbB,
                 float* C, long ldc, long scB, int M, int N, int K,
                 const float* bias, int batch)
{
  dim3 g((unsigned)cdiv(N, TN), (unsigned)cdiv(M, TM), (unsigned)batch);
  if (saK == 1)
    wmma_gemm<1, 1><<<g, 256, 0, st>>>(A, saM, saK, saB, B, sbK, sbN, sbB,
                                       C, ldc, scB, M, N, K, bias);
  else
    wmma_gemm<0, 1><<<g, 256, 0, st>>>(A, saM, saK, saB, B, sbK, sbN, sbB,
                                       C, ldc, scB, M, N, K, bias);
}

float* bn_only(Arena& W, const float* x, const float* g, const float* bt,
               int B, int C, long S, int act, const float* resid, hipStream_t st)
{
  float* mean = W.alloc(C);
  float* rstd = W.alloc(C);
  bn_stats<<<C, 256, 0, st>>>(x, B, C, S, mean, rstd);
  float* y = W.alloc((size_t)B * C * S);
  long sb = cdiv(S, 256); if (sb > 256) sb = 256;
  bn_apply<<<dim3((unsigned)sb, (unsigned)C, (unsigned)B), 256, 0, st>>>(
      x, y, mean, rstd, g, bt, resid, C, S, act);
  return y;
}

float* lin_bn(Arena& W, const float* x, const float* w, const float* bias,
              const float* g, const float* bt, int B, int Cin, int Cout, int Nn,
              int act, const float* resid, hipStream_t st)
{
  float* y = W.alloc((size_t)B * Cout * Nn);
  launch_gemm(st, w, Cin, 1, 0, x, Nn, 1, (long)Cin * Nn,
              y, Nn, (long)Cout * Nn, Cout, Nn, Cin, bias, B);
  return bn_only(W, y, g, bt, B, Cout, Nn, act, resid, st);
}

float* conv_bn(Arena& W, float* col, const float* x, const float* w,
               const float* bias, const float* g, const float* bt,
               int B, int Cin, int H, int Wd, int Cout, int kh, int kw,
               int stride, int pad, int act, const float* resid,
               int& Ho, int& Wo, hipStream_t st)
{
  Ho = (H + 2 * pad - kh) / stride + 1;
  Wo = (Wd + 2 * pad - kw) / stride + 1;
  long S = (long)Ho * Wo;
  long K = (long)Cin * kh * kw;
  im2col_k<<<dim3((unsigned)gblocks(K * S), 1, (unsigned)B), 256, 0, st>>>(
      x, col, Cin, H, Wd, kh, kw, stride, pad, Ho, Wo);
  float* y = W.alloc((size_t)B * Cout * S);
  launch_gemm(st, w, K, 1, 0, col, S, 1, K * S,
              y, S, (long)Cout * S, Cout, (int)S, (int)K, bias, B);
  return bn_only(W, y, g, bt, B, Cout, S, act, resid, st);
}

float* resblock_fwd(Ctx& ctx, Arena& W, float* col, const float* x,
                    int B, int Cin, int Cout, int& H, int& Wd, int stride,
                    hipStream_t st)
{
  // insertion order: w1, bn1(g,b), w2, bn2(g,b), [ws, bns(g,b)]
  const float* w1 = ctx.next();
  const float* bn1g = ctx.next(); const float* bn1b = ctx.next();
  const float* w2 = ctx.next();
  const float* bn2g = ctx.next(); const float* bn2b = ctx.next();
  const float *ws = nullptr, *bnsg = nullptr, *bnsb = nullptr;
  bool has_sc = (stride != 1 || Cin != Cout);
  if (has_sc) { ws = ctx.next(); bnsg = ctx.next(); bnsb = ctx.next(); }

  int Ho, Wo, Ho2, Wo2;
  float* o1 = conv_bn(W, col, x, w1, nullptr, bn1g, bn1b, B, Cin, H, Wd, Cout,
                      3, 3, stride, 1, /*relu*/1, nullptr, Ho, Wo, st);
  const float* sc = x;
  if (has_sc)
    sc = conv_bn(W, col, x, ws, nullptr, bnsg, bnsb, B, Cin, H, Wd, Cout,
                 1, 1, stride, 0, 0, nullptr, Ho2, Wo2, st);
  float* o2 = conv_bn(W, col, o1, w2, nullptr, bn2g, bn2b, B, Cout, Ho, Wo,
                      Cout, 3, 3, 1, 1, /*relu*/1, sc, Ho2, Wo2, st);
  H = Ho; Wd = Wo;
  return o2;
}

struct GRP {
  const float *fc1w, *fc1b, *fc1g, *fc1bb;
  const float *gatW, *a_src, *a_dst, *gbias;
  const float *gbng, *gbnb;
  const float *fc2w, *fc2b, *fc2g, *fc2bb;
};
GRP read_grapher(Ctx& c) {
  GRP p;
  p.fc1w = c.next(); p.fc1b = c.next(); p.fc1g = c.next(); p.fc1bb = c.next();
  p.gatW = c.next(); p.a_src = c.next(); p.a_dst = c.next(); p.gbias = c.next();
  p.gbng = c.next(); p.gbnb = c.next();
  p.fc2w = c.next(); p.fc2b = c.next(); p.fc2g = c.next(); p.fc2bb = c.next();
  return p;
}

struct FFNP { const float *w1, *b1, *bn1g, *bn1b, *w2, *b2, *bn2g, *bn2b; };
FFNP read_ffn(Ctx& c) {
  FFNP p;
  p.w1 = c.next(); p.b1 = c.next(); p.bn1g = c.next(); p.bn1b = c.next();
  p.w2 = c.next(); p.b2 = c.next(); p.bn2g = c.next(); p.bn2b = c.next();
  return p;
}

float* grapher_fwd(const GRP& p, Arena& W, const float* x, int B, int C, int Nn,
                   hipStream_t st)
{
  int D2 = 2 * C, F = 4 * D2;
  // y = bn(fc1(x))                       [B, C, Nn]
  float* y = lin_bn(W, x, p.fc1w, p.fc1b, p.fc1g, p.fc1bb, B, C, C, Nn,
                    0, nullptr, st);
  // Gram G = Y^T Y (A transposed via strides) -> kNN indices
  float* G = W.alloc((size_t)B * Nn * Nn);
  launch_gemm(st, y, 1, Nn, (long)C * Nn, y, Nn, 1, (long)C * Nn,
              G, Nn, (long)Nn * Nn, Nn, Nn, C, nullptr, B);
  int* idx = (int*)W.alloc((size_t)B * Nn * 16);
  knn_topk<<<gblocks((long)B * Nn), 256, 0, st>>>(G, Nn, idx, B * Nn);
  // h = xb @ gatW                        [B, Nn, 4*D2]  (xb = Y^T)
  float* h = W.alloc((size_t)B * Nn * F);
  launch_gemm(st, y, 1, Nn, (long)C * Nn, p.gatW, F, 1, 0,
              h, F, (long)Nn * F, Nn, F, C, nullptr, B);
  // attention scores, softmax-aggregate, head-mean, bias -> g [B, D2, Nn]
  float* as_ = W.alloc((size_t)B * Nn * 4);
  float* ad_ = W.alloc((size_t)B * Nn * 4);
  gat_scores<<<gblocks((long)B * Nn * 4), 256, 0, st>>>(h, p.a_src, p.a_dst,
                                                        as_, ad_, B * Nn * 4, D2);
  float* gout = W.alloc((size_t)B * D2 * Nn);
  gat_aggregate<<<dim3((unsigned)Nn, (unsigned)B), 256, 0, st>>>(
      h, idx, as_, ad_, p.gbias, gout, Nn, D2);
  // gelu(bn(g)); fc2 + bn + residual
  float* gn = bn_only(W, gout, p.gbng, p.gbnb, B, D2, Nn, /*gelu*/2, nullptr, st);
  return lin_bn(W, gn, p.fc2w, p.fc2b, p.fc2g, p.fc2bb, B, D2, C, Nn,
                0, x, st);
}

float* ffn_fwd(const FFNP& p, Arena& W, const float* x, int B, int C, int Nn,
               hipStream_t st)
{
  float* y = lin_bn(W, x, p.w1, p.b1, p.bn1g, p.bn1b, B, C, 4 * C, Nn,
                    /*gelu*/2, nullptr, st);
  return lin_bn(W, y, p.w2, p.b2, p.bn2g, p.bn2b, B, 4 * C, C, Nn,
                0, x, st);
}

} // namespace

extern "C" void kernel_launch(void* const* d_in, const int* in_sizes, int n_in,
                              void* d_out, int out_size, void* d_ws, size_t ws_size,
                              hipStream_t stream)
{
  (void)n_in; (void)out_size;
  const float* x = (const float*)d_in[0];
  int B = in_sizes[0] / (3 * 224 * 224);
  Ctx ctx{d_in, 1};

  // workspace: persistent im2col buffer (max: conv1 147x12544 per image) + two
  // ping-pong arenas (block i+1 temps reuse block i-1's arena; residual-safe)
  size_t colBytes = (((size_t)B * 147 * 12544 * sizeof(float)) + 255) & ~(size_t)255;
  char* base = (char*)d_ws;
  float* col = (float*)base;
  size_t rem = (ws_size > colBytes) ? (ws_size - colBytes) : 0;
  Arena ar0{base + colBytes, 0};
  Arena ar1{base + colBytes + rem / 2, 0};
  Arena* Wk = &ar0; Arena* Wn = &ar1;
  auto swap_arena = [&]() { Arena* t = Wk; Wk = Wn; Wn = t; Wk->reset(); };

  // ---------------- CNN ----------------
  int H = 224, Wd = 224, Cc = 64, Ho, Wo;
  const float* conv1w = ctx.next();
  const float* bn1g = ctx.next(); const float* bn1b = ctx.next();
  float* cur = conv_bn(*Wk, col, x, conv1w, nullptr, bn1g, bn1b,
                       B, 3, H, Wd, 64, 7, 7, 2, 3, /*relu*/1, nullptr, Ho, Wo, stream);
  H = Ho; Wd = Wo;                              // 112x112

  swap_arena();
  int Hp = (H + 2 - 3) / 2 + 1, Wp = (Wd + 2 - 3) / 2 + 1;
  float* pl = Wk->alloc((size_t)B * 64 * Hp * Wp);
  maxpool3x3s2<<<gblocks((long)B * 64 * Hp * Wp), 256, 0, stream>>>(
      cur, pl, B * 64, H, Wd, Hp, Wp);
  cur = pl; H = Hp; Wd = Wp;                    // 56x56

  const int louts[3] = {64, 128, 256};
  const int lstr[3] = {1, 2, 2};
  for (int L = 0; L < 3; ++L)
    for (int blk = 0; blk < 2; ++blk) {
      swap_arena();
      int s = (blk == 0) ? lstr[L] : 1;
      cur = resblock_fwd(ctx, *Wk, col, cur, B, Cc, louts[L], H, Wd, s, stream);
      Cc = louts[L];
    }
  // cur: [B,256,14,14]

  // ---------------- GNN ----------------
  const float* proj_w = ctx.next(); const float* proj_b = ctx.next();
  const float* proj_g = ctx.next(); const float* proj_bb = ctx.next();

  // stage params come before downsample/norm/head in insertion order
  GRP gp[2][2]; FFNP fp[2][2];
  for (int s = 0; s < 2; ++s)
    for (int b = 0; b < 2; ++b) { gp[s][b] = read_grapher(ctx); fp[s][b] = read_ffn(ctx); }
  const float* ds_w = ctx.next(); const float* ds_b = ctx.next();
  const float* ds_g = ctx.next(); const float* ds_bb = ctx.next();
  const float* norm_g = ctx.next(); const float* norm_b = ctx.next();
  const float* head_w = ctx.next(); const float* head_b = ctx.next();

  int Nn = H * Wd, C = 256;
  swap_arena();
  cur = lin_bn(*Wk, cur, proj_w, proj_b, proj_g, proj_bb, B, 256, 256, Nn,
               /*gelu*/2, nullptr, stream);      // 1x1 conv == node linear

  for (int s = 0; s < 2; ++s) {
    for (int b = 0; b < 2; ++b) {
      swap_arena();
      cur = grapher_fwd(gp[s][b], *Wk, cur, B, C, Nn, stream);
      swap_arena();
      cur = ffn_fwd(fp[s][b], *Wk, cur, B, C, Nn, stream);
    }
    if (s == 0) {                                // downsample 256->512, 14->7
      swap_arena();
      cur = conv_bn(*Wk, col, cur, ds_w, ds_b, ds_g, ds_bb,
                    B, 256, H, Wd, 512, 3, 3, 2, 1, 0, nullptr, Ho, Wo, stream);
      H = Ho; Wd = Wo; Nn = H * Wd; C = 512;
    }
  }

  // ---------------- head ----------------
  swap_arena();
  float* v = Wk->alloc((size_t)B * C);
  mean_hw<<<gblocks((long)B * C), 256, 0, stream>>>(cur, v, B * C, Nn);
  float* v2 = bn_only(*Wk, v, norm_g, norm_b, B, C, 1, 0, nullptr, stream);
  head_k<<<B, 256, 0, stream>>>(v2, head_w, head_b, (float*)d_out, C);
}
